// BaselineRNN_50809463112167
// MI455X (gfx1250) — matrix operand order repaired and checked by reading, not compiled, not run on hardware
//
#include <hip/hip_runtime.h>

#define T_STEPS 4096
#define BATCH   32
#define HID     512
#define NWG     32
#define HS      16           // hidden units per WG
#define KT      16           // 512/32 K-tiles
#define THREADS 256

typedef __attribute__((ext_vector_type(16))) __bf16 v16bf;
typedef __attribute__((ext_vector_type(8)))  float  v8f;

__device__ __forceinline__ float sigf(float x) {
  return 1.f / (1.f + __expf(-x));
}
__device__ __forceinline__ float tanhf_fast(float x) {
  float e = __expf(-2.f * x);
  return (1.f - e) / (1.f + e);
}

// Element index inside a 16x32 bf16 WMMA A-operand tile (ISA 7.12.2):
// lane = M + 16*(bit3 of k); packed element e = k%8 (+8 for k>=16).
__device__ __forceinline__ int a_idx(int mtile, int ktile, int M, int kl) {
  int lane = M + (((kl >> 3) & 1) << 4);
  int e = (kl < 16) ? (kl & 7) : (8 + (kl & 7));
  return ((mtile * KT + ktile) * 32 + lane) * 16 + e;
}

__global__ void init_kernel(const float* __restrict__ x0,
                            const float* __restrict__ b_lin,
                            float* __restrict__ out,
                            unsigned* __restrict__ ws_words) {
  int tid = blockIdx.x * blockDim.x + threadIdx.x;
  if (tid < 16448) ws_words[tid] = 0u;  // barrier counter + 2x32KB bf16 h buffers
  if (tid < T_STEPS * BATCH) out[tid] = x0[tid] + b_lin[0];
}

__global__ __launch_bounds__(THREADS, 1)
void lstm_persistent(const float* __restrict__ x0,
                     const float* __restrict__ W_ih,
                     const float* __restrict__ W_hh,
                     const float* __restrict__ b_ih,
                     const float* __restrict__ b_hh,
                     const float* __restrict__ W_lin,
                     float* __restrict__ out,
                     unsigned* __restrict__ counter,
                     __bf16* __restrict__ hbuf0,
                     __bf16* __restrict__ hbuf1) {
  // Resident state in LDS (~108 KB of 320 KB per WGP)
  __shared__ __align__(32) __bf16 W_lds[4 * KT * 32 * 16];   // 64 KB, B-operand swizzle
  __shared__ __align__(32) __bf16 hA_lds[2 * KT * 32 * 16];  // 32 KB, A-operand swizzle
  __shared__ float  gates_lds[64 * 32];                      // 8 KB
  __shared__ float  c_lds[BATCH * HS];                       // persistent cell state
  __shared__ float  hf_lds[BATCH * HS];
  __shared__ float  wih_l[64], bsum_l[64], wlin_l[HS];

  const int wg   = blockIdx.x;
  const int tid  = threadIdx.x;
  const int wave = tid >> 5;
  const int lane = tid & 31;

  // ---- one-time staging: bf16 W_hh slice, swizzled as WMMA B operand ----
  for (int i = tid; i < 4 * 16 * HID; i += THREADS) {  // 64 gate rows x 512
    int rowl = i >> 9;           // local gate-row 0..63 (= gate*16 + n)
    int k    = i & 511;
    int gate = rowl >> 4, n = rowl & 15;
    int grow = gate * HID + wg * HS + n;
    // B[K=k][N=n] = W_hh[row][k]; lane = n + 16*(bit3 of k%32), packed element = k%8 (+8 for k%32>=16)
    W_lds[(((gate * KT) + (k >> 5)) * 32 + n + (((k >> 3) & 1) << 4)) * 16 + (((k & 31) < 16) ? (k & 7) : (8 + (k & 7)))] = (__bf16)W_hh[grow * HID + k];
  }
  if (tid < 64) {
    int gate = tid >> 4, n = tid & 15;
    int grow = gate * HID + wg * HS + n;
    wih_l[tid]  = W_ih[grow];
    bsum_l[tid] = b_ih[grow] + b_hh[grow];
  }
  if (tid < HS) wlin_l[tid] = W_lin[wg * HS + tid];
  for (int i = tid; i < BATCH * HS; i += THREADS) c_lds[i] = 0.f;
  __syncthreads();

  const int nt = wave >> 1;  // gate tile 0..3 (i,f,g,o)
  const int mt = wave & 1;   // batch tile 0..1

  // per-thread constants for the async h staging (16 B per lane per issue)
  const unsigned lds_h = (unsigned)(size_t)(&hA_lds[0]) + tid * 16;
  const unsigned voff  = tid * 16;

  for (int t = 0; t < T_STEPS; ++t) {
    const __bf16* hcur = (t & 1) ? hbuf1 : hbuf0;
    __bf16*       hnxt = (t & 1) ? hbuf0 : hbuf1;

    // stage h_{t-1} (bf16, pre-swizzled in global/L2) into LDS: 32 KB via the
    // CDNA5 async global->LDS path (no VGPR round trip, ASYNCcnt tracked).
    {
      const unsigned long long gb = (unsigned long long)hcur;
      asm volatile("global_load_async_to_lds_b128 %0, %1, %2 offset:0"
                   :: "v"(lds_h), "v"(voff), "s"(gb) : "memory");
      asm volatile("global_load_async_to_lds_b128 %0, %1, %2 offset:4096"
                   :: "v"(lds_h), "v"(voff), "s"(gb) : "memory");
      asm volatile("global_load_async_to_lds_b128 %0, %1, %2 offset:8192"
                   :: "v"(lds_h), "v"(voff), "s"(gb) : "memory");
      asm volatile("global_load_async_to_lds_b128 %0, %1, %2 offset:12288"
                   :: "v"(lds_h), "v"(voff), "s"(gb) : "memory");
      asm volatile("global_load_async_to_lds_b128 %0, %1, %2 offset:16384"
                   :: "v"(lds_h), "v"(voff), "s"(gb) : "memory");
      asm volatile("global_load_async_to_lds_b128 %0, %1, %2 offset:20480"
                   :: "v"(lds_h), "v"(voff), "s"(gb) : "memory");
      asm volatile("global_load_async_to_lds_b128 %0, %1, %2 offset:24576"
                   :: "v"(lds_h), "v"(voff), "s"(gb) : "memory");
      asm volatile("global_load_async_to_lds_b128 %0, %1, %2 offset:28672"
                   :: "v"(lds_h), "v"(voff), "s"(gb) : "memory");
    }
    // warm L2/L0 for next step's x0 row while the async copy is in flight
    if (tid == 0 && t + 1 < T_STEPS) __builtin_prefetch(&x0[(t + 1) * BATCH], 0, 0);
    __builtin_amdgcn_s_wait_asynccnt(0);
    __syncthreads();

    // gates tile = h @ W_hh^T : one 16x16 f32 tile per wave, K=512
    v8f acc = {};
    #pragma unroll
    for (int kt = 0; kt < KT; ++kt) {
      v16bf a = *(const v16bf*)&hA_lds[((mt * KT + kt) * 32 + lane) * 16];
      v16bf b = *(const v16bf*)&W_lds[((nt * KT + kt) * 32 + lane) * 16];
      acc = __builtin_amdgcn_wmma_f32_16x16x32_bf16(false, a, false, b,
                                                    (short)0, acc, false, false);
    }
    {
      int n = lane & 15;
      #pragma unroll
      for (int i = 0; i < 8; ++i) {
        int M = (lane < 16) ? i : (i + 8);
        gates_lds[(nt * 16 + n) * 32 + (mt * 16 + M)] = acc[i];
      }
    }
    __syncthreads();

    // pointwise LSTM cell update: 32 batch x 16 hidden per WG
    #pragma unroll
    for (int r = 0; r < 2; ++r) {
      int idx = tid + r * THREADS;  // = b*16 + n
      int b_  = idx >> 4;
      int n   = idx & 15;
      float xv = x0[t * BATCH + b_];
      float gi = gates_lds[(n)      * 32 + b_] + xv * wih_l[n]      + bsum_l[n];
      float gf = gates_lds[(16 + n) * 32 + b_] + xv * wih_l[16 + n] + bsum_l[16 + n];
      float gg = gates_lds[(32 + n) * 32 + b_] + xv * wih_l[32 + n] + bsum_l[32 + n];
      float go = gates_lds[(48 + n) * 32 + b_] + xv * wih_l[48 + n] + bsum_l[48 + n];
      float iv = sigf(gi), fv = sigf(gf), gv = tanhf_fast(gg), ov = sigf(go);
      float c = fv * c_lds[idx] + iv * gv;
      c_lds[idx] = c;
      float h = ov * tanhf_fast(c);
      hf_lds[idx] = h;
      int jg = wg * HS + n;  // global hidden index = K position for next step
      hnxt[a_idx(b_ >> 4, jg >> 5, b_ & 15, jg & 31)] = (__bf16)h;
    }
    __syncthreads();

    // partial output projection for this WG's hidden slice
    if (tid < BATCH) {
      float s = 0.f;
      #pragma unroll
      for (int n = 0; n < HS; ++n) s += hf_lds[tid * HS + n] * wlin_l[n];
      atomicAdd(&out[t * BATCH + tid], s);
    }

    // device-wide barrier (monotonic counter, one per timestep)
    __syncthreads();
    if (tid == 0) {
      __threadfence();
      atomicAdd(counter, 1u);
      unsigned target = (unsigned)(t + 1) * NWG;
      while (__hip_atomic_load(counter, __ATOMIC_ACQUIRE, __HIP_MEMORY_SCOPE_AGENT) < target) {
        __builtin_amdgcn_s_sleep(2);
      }
    }
    __syncthreads();
  }
}

extern "C" void kernel_launch(void* const* d_in, const int* in_sizes, int n_in,
                              void* d_out, int out_size, void* d_ws, size_t ws_size,
                              hipStream_t stream) {
  const float* x0    = (const float*)d_in[0];
  const float* W_ih  = (const float*)d_in[1];
  const float* W_hh  = (const float*)d_in[2];
  const float* b_ih  = (const float*)d_in[3];
  const float* b_hh  = (const float*)d_in[4];
  const float* W_lin = (const float*)d_in[5];
  const float* b_lin = (const float*)d_in[6];
  float* out = (float*)d_out;

  unsigned* ws_words = (unsigned*)d_ws;
  unsigned* counter  = ws_words;                              // word 0
  __bf16* hbuf0 = (__bf16*)((char*)d_ws + 256);               // 32 KB
  __bf16* hbuf1 = (__bf16*)((char*)d_ws + 256 + 32 * 1024);   // 32 KB

  init_kernel<<<(T_STEPS * BATCH + 255) / 256, 256, 0, stream>>>(x0, b_lin, out, ws_words);
  lstm_persistent<<<NWG, THREADS, 0, stream>>>(x0, W_ih, W_hh, b_ih, b_hh, W_lin,
                                               out, counter, hbuf0, hbuf1);
}
